// RWKV_532575945179
// MI455X (gfx1250) — compile-verified, matrix operand order checked by reading, not run on hardware
//
#include <hip/hip_runtime.h>

// ---------------------------------------------------------------------------
// RWKV-7 forward for MI455X (gfx1250, wave32, WMMA).
// All dense projections run as bf16 WMMA GEMMs (v_wmma_f32_16x16x32_bf16)
// with fp32 accumulation; the 24 sequential state scans run as one WG each.
// x (residual stream) lives in d_out; GEMM epilogue mode 1 accumulates into it.
// ---------------------------------------------------------------------------

typedef __bf16 bf16;
typedef __attribute__((ext_vector_type(16))) bf16  v16bf;
typedef __attribute__((ext_vector_type(8)))  bf16  v8bf;
typedef __attribute__((ext_vector_type(8)))  float v8f;

#define Bq 2
#define Tq 1024
#define Cq 768
#define Hq 12
#define Nq 64
#define Lq 6
#define FFq 3072
#define Mq (Bq*Tq)          // 2048 rows of activations

__device__ __forceinline__ float wave_sum(float v) {
  #pragma unroll
  for (int off = 16; off > 0; off >>= 1) v += __shfl_xor(v, off, 32);
  return v;
}

// ---------------- embedding gather ----------------
__global__ void k_embed(const int* __restrict__ idx, const float* __restrict__ emb,
                        float* __restrict__ x) {
  size_t i = (size_t)blockIdx.x * blockDim.x + threadIdx.x;
  if (i >= (size_t)Mq * Cq) return;
  int tok = (int)(i / Cq), c = (int)(i % Cq);
  x[i] = emb[(size_t)idx[tok] * Cq + c];
}

// ---------------- layernorm (one WG per token) ----------------
__global__ __launch_bounds__(256)
void k_ln(const float* __restrict__ in, const float* __restrict__ w,
          const float* __restrict__ b, float* __restrict__ out) {
  __shared__ float red[256];
  const int t = blockIdx.x;
  const float* row = in + (size_t)t * Cq;
  float s = 0.f;
  for (int c = threadIdx.x; c < Cq; c += 256) s += row[c];
  red[threadIdx.x] = s; __syncthreads();
  for (int o = 128; o > 0; o >>= 1) {
    if (threadIdx.x < o) red[threadIdx.x] += red[threadIdx.x + o];
    __syncthreads();
  }
  float mean = red[0] * (1.f / Cq);
  __syncthreads();
  float s2 = 0.f;
  for (int c = threadIdx.x; c < Cq; c += 256) { float d = row[c] - mean; s2 += d * d; }
  red[threadIdx.x] = s2; __syncthreads();
  for (int o = 128; o > 0; o >>= 1) {
    if (threadIdx.x < o) red[threadIdx.x] += red[threadIdx.x + o];
    __syncthreads();
  }
  float rstd = rsqrtf(red[0] * (1.f / Cq) + 1e-5f);
  for (int c = threadIdx.x; c < Cq; c += 256)
    out[(size_t)t * Cq + c] = (row[c] - mean) * rstd * w[c] + b[c];
}

// ---------------- time-shift mixes (tmix: 6 outputs, bf16) ----------------
__global__ void k_shiftmix6(const float* __restrict__ xn,
                            const float* pr, const float* pw, const float* pk,
                            const float* pv, const float* pa, const float* pg,
                            bf16* orr, bf16* ow, bf16* ok, bf16* ov, bf16* oa, bf16* og) {
  size_t i = (size_t)blockIdx.x * blockDim.x + threadIdx.x;
  if (i >= (size_t)Mq * Cq) return;
  int c = (int)(i % Cq);
  int tok = (int)(i / Cq);
  float cur = xn[i];
  float prev = (tok % Tq) > 0 ? xn[i - Cq] : 0.f;
  float xx = prev - cur;
  orr[i] = (bf16)(cur + xx * pr[c]);
  ow[i]  = (bf16)(cur + xx * pw[c]);
  ok[i]  = (bf16)(cur + xx * pk[c]);
  ov[i]  = (bf16)(cur + xx * pv[c]);
  oa[i]  = (bf16)(cur + xx * pa[c]);
  og[i]  = (bf16)(cur + xx * pg[c]);
}

__global__ void k_shiftmix1(const float* __restrict__ xn, const float* pk,
                            bf16* __restrict__ ok) {
  size_t i = (size_t)blockIdx.x * blockDim.x + threadIdx.x;
  if (i >= (size_t)Mq * Cq) return;
  int c = (int)(i % Cq);
  int tok = (int)(i / Cq);
  float cur = xn[i];
  float prev = (tok % Tq) > 0 ? xn[i - Cq] : 0.f;
  ok[i] = (bf16)(cur + (prev - cur) * pk[c]);
}

// ---------------- weight transpose + bf16 convert: W[K,N] -> WT[N,K] -------
__global__ void k_convT(const float* __restrict__ W, bf16* __restrict__ WT,
                        int K, int Nd) {
  size_t i = (size_t)blockIdx.x * blockDim.x + threadIdx.x;
  if (i >= (size_t)K * Nd) return;
  int k = (int)(i / Nd), n = (int)(i % Nd);
  WT[(size_t)n * K + k] = (bf16)W[i];
}

// ---------------- WMMA GEMM: C[M,Nd] = A[M,K](bf16) x W[K,Nd] ----------------
// W supplied transposed: BT[Nd,K] bf16, so both fragments are contiguous.
// 128x128 WG tile, 8 waves, each 32x64 via 2x4 16x16x32 bf16 WMMAs.
// mode: 0 f32 store, 1 f32 add-to-dest, 2 relu^2->bf16, 3 bf16,
//       4 tanh->bf16, 5 sigmoid->f32, 6 sigmoid->bf16
__global__ __launch_bounds__(256)
void k_gemm(const bf16* __restrict__ A, const bf16* __restrict__ BT,
            const float* __restrict__ bias, float* __restrict__ outF,
            bf16* __restrict__ outB, int M, int K, int Nd, int mode) {
  __shared__ bf16 sA[128 * 32];
  __shared__ bf16 sB[128 * 32];
  const int tid  = threadIdx.x;
  const int lane = tid & 31;
  const int wid  = tid >> 5;
  const int wm   = wid & 3;      // 4 wave-rows of 32
  const int wn   = wid >> 2;     // 2 wave-cols of 64
  const int bm   = blockIdx.x * 128;
  const int bn   = blockIdx.y * 128;
  const int frow = lane & 15;
  const int kh   = (lane >> 4) * 8;   // bf16 A 16x32 lane layout: K base 0/8

  v8f acc[2][4];
  #pragma unroll
  for (int i = 0; i < 2; ++i)
    #pragma unroll
    for (int j = 0; j < 4; ++j) acc[i][j] = (v8f){};

  const int ar   = tid >> 1;           // staging row 0..127
  const int aseg = (tid & 1) * 16;     // 16 bf16 per thread

  for (int k0 = 0; k0 < K; k0 += 32) {
    { // stage A tile (rows always in range: M % 128 == 0)
      const bf16* gp = A + (size_t)(bm + ar) * K + k0 + aseg;
      if (k0 + 32 < K) __builtin_prefetch(gp + 32, 0, 1);  // global_prefetch_b8
      v8bf p0 = *(const v8bf*)gp;
      v8bf p1 = *(const v8bf*)(gp + 8);
      *(v8bf*)&sA[ar * 32 + aseg]     = p0;
      *(v8bf*)&sA[ar * 32 + aseg + 8] = p1;
    }
    { // stage BT tile (guard Nd < 128)
      int br = bn + ar;
      v8bf p0{}, p1{};
      if (br < Nd) {
        const bf16* gp = BT + (size_t)br * K + k0 + aseg;
        p0 = *(const v8bf*)gp;
        p1 = *(const v8bf*)(gp + 8);
      }
      *(v8bf*)&sB[ar * 32 + aseg]     = p0;
      *(v8bf*)&sB[ar * 32 + aseg + 8] = p1;
    }
    __syncthreads();

    union Frag { v16bf v; v8bf h[2]; };
    Frag fa[2], fb[4];
    #pragma unroll
    for (int i = 0; i < 2; ++i) {
      int r = wm * 32 + i * 16 + frow;
      fa[i].h[0] = *(const v8bf*)&sA[r * 32 + kh];
      fa[i].h[1] = *(const v8bf*)&sA[r * 32 + kh + 16];
    }
    #pragma unroll
    for (int j = 0; j < 4; ++j) {
      int c = wn * 64 + j * 16 + frow;
      fb[j].h[0] = *(const v8bf*)&sB[c * 32 + kh];
      fb[j].h[1] = *(const v8bf*)&sB[c * 32 + kh + 16];
    }
    #pragma unroll
    for (int i = 0; i < 2; ++i)
      #pragma unroll
      for (int j = 0; j < 4; ++j)
        acc[i][j] = __builtin_amdgcn_wmma_f32_16x16x32_bf16(
            false, fa[i].v, false, fb[j].v, (short)0, acc[i][j], false, false);
    __syncthreads();
  }

  // epilogue: C/D layout -> n = lane&15, m = 8*(lane>>4) + vgpr index
  #pragma unroll
  for (int i = 0; i < 2; ++i) {
    #pragma unroll
    for (int j = 0; j < 4; ++j) {
      int n = bn + wn * 64 + j * 16 + (lane & 15);
      if (n >= Nd) continue;
      #pragma unroll
      for (int rr = 0; rr < 8; ++rr) {
        int m = bm + wm * 32 + i * 16 + (lane >> 4) * 8 + rr;
        float val = acc[i][j][rr];
        if (bias) val += bias[n];
        size_t off = (size_t)m * Nd + n;
        switch (mode) {
          case 0: outF[off] = val; break;
          case 1: outF[off] += val; break;
          case 2: { float t = val > 0.f ? val : 0.f; outB[off] = (bf16)(t * t); } break;
          case 3: outB[off] = (bf16)val; break;
          case 4: outB[off] = (bf16)tanhf(val); break;
          case 5: outF[off] = 1.f / (1.f + __expf(-val)); break;
          case 6: outB[off] = (bf16)(1.f / (1.f + __expf(-val))); break;
        }
      }
    }
  }
}

// ---------------- decay finalize: wdec = exp(-exp(-softplus(-y)-0.5)) ------
__global__ void k_wfinal(const float* __restrict__ pre, float* __restrict__ wdec) {
  size_t i = (size_t)blockIdx.x * blockDim.x + threadIdx.x;
  if (i >= (size_t)Mq * Cq) return;
  float y = pre[i];
  float wraw = -log1pf(__expf(-y)) - 0.5f;
  wdec[i] = __expf(-__expf(wraw));
}

// ---------------- v_first blend ----------------
__global__ void k_vmix(const float* __restrict__ v0, const float* __restrict__ vsig,
                       float* __restrict__ v, float* __restrict__ vfirst, int layer) {
  size_t i = (size_t)blockIdx.x * blockDim.x + threadIdx.x;
  if (i >= (size_t)Mq * Cq) return;
  if (layer == 0) { v[i] = v0[i]; vfirst[i] = v0[i]; }
  else            { v[i] = v0[i] + (vfirst[i] - v0[i]) * vsig[i]; }
}

// ---------------- per-(token,head): kk normalize, k adjust -----------------
__global__ __launch_bounds__(256)
void k_prep(const float* __restrict__ k, const float* __restrict__ a,
            const float* __restrict__ kkp, const float* __restrict__ kap,
            float* __restrict__ k2, float* __restrict__ aA, float* __restrict__ bB) {
  int wave = blockIdx.x * 8 + (threadIdx.x >> 5);
  int lane = threadIdx.x & 31;
  int tok = wave / Hq, h = wave % Hq;
  size_t base = (size_t)tok * Cq + h * Nq;
  int n0 = lane, n1 = lane + 32;
  int c0 = h * Nq + n0, c1 = h * Nq + n1;
  float k0v = k[base + n0], k1v = k[base + n1];
  float a0v = a[base + n0], a1v = a[base + n1];
  float kk0 = k0v * kkp[c0], kk1 = k1v * kkp[c1];
  float nrm = sqrtf(wave_sum(kk0 * kk0 + kk1 * kk1));
  nrm = nrm < 1e-12f ? 1e-12f : nrm;
  float kn0 = kk0 / nrm, kn1 = kk1 / nrm;
  aA[base + n0] = -kn0;       aA[base + n1] = -kn1;
  bB[base + n0] = kn0 * a0v;  bB[base + n1] = kn1 * a1v;
  k2[base + n0] = k0v * (1.f + (a0v - 1.f) * kap[c0]);
  k2[base + n1] = k1v * (1.f + (a1v - 1.f) * kap[c1]);
}

// ---------------- RWKV-7 state scan: one WG per (b,h) ----------------------
__global__ __launch_bounds__(64)
void k_rwkv(const float* __restrict__ r, const float* __restrict__ w,
            const float* __restrict__ kx, const float* __restrict__ v,
            const float* __restrict__ aA, const float* __restrict__ bB,
            float* __restrict__ o) {
  const int bh = blockIdx.x;          // 0..23
  const int b = bh / Hq, h = bh % Hq;
  const int i = threadIdx.x;          // state row 0..63
  __shared__ float sh[6][Nq];
  float st[Nq];
  #pragma unroll
  for (int j = 0; j < Nq; ++j) st[j] = 0.f;

  for (int t = 0; t < Tq; ++t) {
    size_t base = ((size_t)(b * Tq + t)) * Cq + h * Nq;
    sh[0][i] = r[base + i];  sh[1][i] = w[base + i];  sh[2][i] = kx[base + i];
    sh[3][i] = v[base + i];  sh[4][i] = aA[base + i]; sh[5][i] = bB[base + i];
    __syncthreads();
    float sa = 0.f;
    #pragma unroll
    for (int j = 0; j < Nq; ++j) sa += st[j] * sh[4][j];
    float vi = sh[3][i];
    float out = 0.f;
    #pragma unroll
    for (int j = 0; j < Nq; ++j) {
      float s = st[j] * sh[1][j] + sa * sh[5][j] + vi * sh[2][j];
      st[j] = s;
      out += s * sh[0][j];
    }
    o[base + i] = out;
    __syncthreads();
  }
}

// ---------------- groupnorm + bonus + gate, bf16 out ------------------------
__global__ __launch_bounds__(256)
void k_gnbonus(const float* __restrict__ o, const float* __restrict__ r,
               const float* __restrict__ k2, const float* __restrict__ v,
               const float* __restrict__ g, const float* __restrict__ gnw,
               const float* __restrict__ gnb, const float* __restrict__ rk,
               bf16* __restrict__ og) {
  int wave = blockIdx.x * 8 + (threadIdx.x >> 5);
  int lane = threadIdx.x & 31;
  int tok = wave / Hq, h = wave % Hq;
  size_t base = (size_t)tok * Cq + h * Nq;
  int n0 = lane, n1 = lane + 32;
  int c0 = h * Nq + n0, c1 = h * Nq + n1;
  float o0 = o[base + n0], o1 = o[base + n1];
  float m = wave_sum(o0 + o1) * (1.f / Nq);
  float d0 = o0 - m, d1 = o1 - m;
  float var = wave_sum(d0 * d0 + d1 * d1) * (1.f / Nq);
  float rstd = rsqrtf(var + 0.00064f);
  float s = wave_sum(r[base + n0] * k2[base + n0] * rk[c0] +
                     r[base + n1] * k2[base + n1] * rk[c1]);
  float y0 = d0 * rstd * gnw[c0] + gnb[c0] + s * v[base + n0];
  float y1 = d1 * rstd * gnw[c1] + gnb[c1] + s * v[base + n1];
  og[base + n0] = (bf16)(y0 * g[base + n0]);
  og[base + n1] = (bf16)(y1 * g[base + n1]);
}

// ===========================================================================
extern "C" void kernel_launch(void* const* d_in, const int* in_sizes, int n_in,
                              void* d_out, int out_size, void* d_ws, size_t ws_size,
                              hipStream_t stream) {
  (void)in_sizes; (void)n_in; (void)out_size; (void)ws_size;
  const int*   idx    = (const int*)  d_in[0];
  const float* emb    = (const float*)d_in[1];
  const float* ln0_w  = (const float*)d_in[2];
  const float* ln0_b  = (const float*)d_in[3];
  const float* ln1_w  = (const float*)d_in[4];
  const float* ln1_b  = (const float*)d_in[5];
  const float* ln2_w  = (const float*)d_in[6];
  const float* ln2_b  = (const float*)d_in[7];
  const float* x_r    = (const float*)d_in[8];
  const float* x_w    = (const float*)d_in[9];
  const float* x_k    = (const float*)d_in[10];
  const float* x_v    = (const float*)d_in[11];
  const float* x_a    = (const float*)d_in[12];
  const float* x_g    = (const float*)d_in[13];
  const float* w0     = (const float*)d_in[14];
  const float* w1     = (const float*)d_in[15];
  const float* w2     = (const float*)d_in[16];
  const float* a0     = (const float*)d_in[17];
  const float* a1     = (const float*)d_in[18];
  const float* a2     = (const float*)d_in[19];
  const float* v0p    = (const float*)d_in[20];
  const float* v1     = (const float*)d_in[21];
  const float* v2     = (const float*)d_in[22];
  const float* g1     = (const float*)d_in[23];
  const float* g2     = (const float*)d_in[24];
  const float* k_k    = (const float*)d_in[25];
  const float* k_a    = (const float*)d_in[26];
  const float* r_k    = (const float*)d_in[27];
  const float* Wr     = (const float*)d_in[28];
  const float* Wk     = (const float*)d_in[29];
  const float* Wv     = (const float*)d_in[30];
  const float* Wo     = (const float*)d_in[31];
  const float* gn_w   = (const float*)d_in[32];
  const float* gn_b   = (const float*)d_in[33];
  const float* fx_k   = (const float*)d_in[34];
  const float* Wkey   = (const float*)d_in[35];
  const float* Wval   = (const float*)d_in[36];

  // workspace layout
  const size_t SZ = (size_t)Mq * Cq;        // 1,572,864 elems
  const size_t F  = SZ * 4;                 // f32 bytes
  const size_t Hb = SZ * 2;                 // bf16 bytes
  char* ws = (char*)d_ws;
  size_t off = 0;
  auto takeF = [&](void) { float* p = (float*)(ws + off); off += F;  return p; };
  auto takeB = [&](size_t bytes) { bf16* p = (bf16*)(ws + off); off += bytes; return p; };

  float* xn    = takeF();
  bf16*  mixR  = takeB(Hb); bf16* mixW = takeB(Hb); bf16* mixK = takeB(Hb);
  bf16*  mixV  = takeB(Hb); bf16* mixA = takeB(Hb); bf16* mixG = takeB(Hb);
  float* rbuf  = takeF();
  float* wdec  = takeF();
  float* kbuf  = takeF();
  float* vbuf  = takeF();
  float* v0buf = takeF();
  float* vfirst= takeF();
  float* abuf  = takeF();
  float* gbuf  = takeF();
  float* tmp   = takeF();
  float* k2buf = takeF();
  float* aAbuf = takeF();
  float* bBbuf = takeF();
  float* obuf  = takeF();
  bf16*  ogbuf = takeB(Hb);
  bf16*  kxbuf = takeB(Hb);
  bf16*  tsbuf = takeB((size_t)Mq * 128 * 2);
  bf16*  hbuf  = takeB((size_t)Mq * FFq * 2);
  bf16*  WT    = takeB((size_t)Cq * FFq * 2);

  float* x = (float*)d_out;                  // residual stream = output

  const int EW  = (int)((SZ + 255) / 256);   // elementwise grid
  const int PW  = (Mq * Hq) / 8;             // per-(token,head) wave grid

  // convert-transpose + GEMM helper
  auto gemm = [&](const bf16* Abuf, const float* W, int K, int Nd,
                  const float* bias, float* oF, bf16* oB, int mode) {
    int total = K * Nd;
    k_convT<<<(total + 255) / 256, 256, 0, stream>>>(W, WT, K, Nd);
    dim3 g(Mq / 128, (Nd + 127) / 128);
    k_gemm<<<g, 256, 0, stream>>>(Abuf, WT, bias, oF, oB, Mq, K, Nd, mode);
  };

  // embedding + ln0 (in place on x)
  k_embed<<<EW, 256, 0, stream>>>(idx, emb, x);
  k_ln<<<Mq, 256, 0, stream>>>(x, ln0_w, ln0_b, x);

  for (int i = 0; i < Lq; ++i) {
    const size_t cO = (size_t)i * Cq;
    // ---- tmix ----
    k_ln<<<Mq, 256, 0, stream>>>(x, ln1_w + cO, ln1_b + cO, xn);
    k_shiftmix6<<<EW, 256, 0, stream>>>(xn, x_r + cO, x_w + cO, x_k + cO,
                                        x_v + cO, x_a + cO, x_g + cO,
                                        mixR, mixW, mixK, mixV, mixA, mixG);
    gemm(mixR, Wr + (size_t)i * Cq * Cq, Cq, Cq, nullptr, rbuf, nullptr, 0);
    gemm(mixK, Wk + (size_t)i * Cq * Cq, Cq, Cq, nullptr, kbuf, nullptr, 0);
    gemm(mixV, Wv + (size_t)i * Cq * Cq, Cq, Cq, nullptr, v0buf, nullptr, 0);
    // decay LoRA: tanh(xw@w1)@w2 + w0 -> softplus/exp chain
    gemm(mixW, w1 + (size_t)i * Cq * 64, Cq, 64, nullptr, nullptr, tsbuf, 4);
    gemm(tsbuf, w2 + (size_t)i * 64 * Cq, 64, Cq, w0 + cO, tmp, nullptr, 0);
    k_wfinal<<<EW, 256, 0, stream>>>(tmp, wdec);
    // v blend LoRA
    gemm(mixV, v1 + (size_t)i * Cq * 32, Cq, 32, nullptr, nullptr, tsbuf, 3);
    gemm(tsbuf, v2 + (size_t)i * 32 * Cq, 32, Cq, v0p + cO, tmp, nullptr, 5);
    k_vmix<<<EW, 256, 0, stream>>>(v0buf, tmp, vbuf, vfirst, i);
    // a LoRA
    gemm(mixA, a1 + (size_t)i * Cq * 64, Cq, 64, nullptr, nullptr, tsbuf, 3);
    gemm(tsbuf, a2 + (size_t)i * 64 * Cq, 64, Cq, a0 + cO, abuf, nullptr, 5);
    // gate LoRA
    gemm(mixG, g1 + (size_t)i * Cq * 128, Cq, 128, nullptr, nullptr, tsbuf, 6);
    gemm(tsbuf, g2 + (size_t)i * 128 * Cq, 128, Cq, nullptr, gbuf, nullptr, 0);
    // kk normalize / k adjust, scan, groupnorm+bonus+gate
    k_prep<<<PW, 256, 0, stream>>>(kbuf, abuf, k_k + cO, k_a + cO,
                                   k2buf, aAbuf, bBbuf);
    k_rwkv<<<Bq * Hq, 64, 0, stream>>>(rbuf, wdec, k2buf, vbuf, aAbuf, bBbuf, obuf);
    k_gnbonus<<<PW, 256, 0, stream>>>(obuf, rbuf, k2buf, vbuf, gbuf,
                                      gn_w + cO, gn_b + cO, r_k + cO, ogbuf);
    // output projection, residual add into x
    gemm(ogbuf, Wo + (size_t)i * Cq * Cq, Cq, Cq, nullptr, x, nullptr, 1);
    // ---- cmix ----
    k_ln<<<Mq, 256, 0, stream>>>(x, ln2_w + cO, ln2_b + cO, xn);
    k_shiftmix1<<<EW, 256, 0, stream>>>(xn, fx_k + cO, kxbuf);
    gemm(kxbuf, Wkey + (size_t)i * Cq * FFq, Cq, FFq, nullptr, nullptr, hbuf, 2);
    gemm(hbuf, Wval + (size_t)i * FFq * Cq, FFq, Cq, nullptr, x, nullptr, 1);
  }
}
